// LSTM_12146167513194
// MI455X (gfx1250) — matrix operand order repaired and checked by reading, not compiled, not run on hardware
//
#include <hip/hip_runtime.h>
#include <math.h>

#define B_ 128
#define S_ 1024
#define D_ 256
#define H_ 512
#define C_ 10
#define KTOT (D_ + H_) /* 768 */
#define NBLK 256       /* 8 x 32 persistent workgroups */

typedef __bf16 bf16_t;
typedef __attribute__((ext_vector_type(16))) __bf16 v16bf;
typedef __attribute__((ext_vector_type(8)))  float  v8f;
typedef __attribute__((ext_vector_type(4)))  float  f4;
typedef __attribute__((ext_vector_type(4)))  int    v4i;

#define AS1 __attribute__((address_space(1)))
#define AS3 __attribute__((address_space(3)))

union Frag {
    v16bf v;
    f4    q[2];
};

__device__ __forceinline__ bf16_t f2bf(float f) {
    unsigned u = __float_as_uint(f);
    u += 0x7FFFu + ((u >> 16) & 1u);           // round-to-nearest-even
    unsigned short s = (unsigned short)(u >> 16);
    return __builtin_bit_cast(bf16_t, s);
}
__device__ __forceinline__ float bf2f(bf16_t b) {
    unsigned short s = __builtin_bit_cast(unsigned short, b);
    return __uint_as_float(((unsigned)s) << 16);
}

// ---- CDNA5 async global->LDS copy (ASYNCcnt path) -------------------------
__device__ __forceinline__ void async_load_b128(const bf16_t* g, bf16_t* l) {
#if __has_builtin(__builtin_amdgcn_global_load_async_to_lds_b128)
    __builtin_amdgcn_global_load_async_to_lds_b128(
        (AS1 v4i*)(unsigned long long)g, (AS3 v4i*)l, 0, 0);
#else
    asm volatile("global_load_async_to_lds_b128 %0, %1, off"
                 :: "v"((AS3 v4i*)l), "v"(g) : "memory");
#endif
}
__device__ __forceinline__ void wait_async0() {
#if __has_builtin(__builtin_amdgcn_s_wait_asynccnt)
    __builtin_amdgcn_s_wait_asynccnt(0);
#else
    asm volatile("s_wait_asynccnt 0x0" ::: "memory");
#endif
}

__device__ __forceinline__ void load_frags(Frag& a, Frag& b,
                                           const bf16_t* aP, const bf16_t* bP, int k) {
    a.q[0] = *(const f4*)(aP + k);
    a.q[1] = *(const f4*)(aP + k + 16);
    b.q[0] = *(const f4*)(bP + k);
    b.q[1] = *(const f4*)(bP + k + 16);
}

// ---- convert x [B,S,D] fp32 -> bf16 --------------------------------------
__global__ void conv_x_kernel(const float* __restrict__ x, bf16_t* __restrict__ xbf, int n) {
    int i = (blockIdx.x * blockDim.x + threadIdx.x) * 4;
    if (i + 3 < n) {
        f4 v = *(const f4*)(x + i);
        xbf[i + 0] = f2bf(v[0]);
        xbf[i + 1] = f2bf(v[1]);
        xbf[i + 2] = f2bf(v[2]);
        xbf[i + 3] = f2bf(v[3]);
    }
}

// ---- pack weights: wpack[g][n][k] = (k<D ? Wx[k][n] : Wh[k-D][n]) in bf16 -
__global__ void pack_w_kernel(const float* __restrict__ Wgx, const float* __restrict__ Wix,
                              const float* __restrict__ Wfx, const float* __restrict__ Wox,
                              const float* __restrict__ Wgh, const float* __restrict__ Wih,
                              const float* __restrict__ Wfh, const float* __restrict__ Woh,
                              bf16_t* __restrict__ wpack) {
    int gid = blockIdx.x * blockDim.x + threadIdx.x;
    if (gid >= 4 * H_ * KTOT) return;
    int g   = gid / (H_ * KTOT);
    int rem = gid % (H_ * KTOT);
    int n   = rem / KTOT;
    int k   = rem % KTOT;
    const float* Wx = (g == 0) ? Wgx : (g == 1) ? Wix : (g == 2) ? Wfx : Wox;
    const float* Wh = (g == 0) ? Wgh : (g == 1) ? Wih : (g == 2) ? Wfh : Woh;
    float v = (k < D_) ? Wx[(size_t)k * H_ + n] : Wh[(size_t)(k - D_) * H_ + n];
    wpack[gid] = f2bf(v);
}

// ---- zero h0/h1 (bf16), c (f32), sync counter ----------------------------
__global__ void init_state_kernel(bf16_t* __restrict__ h0, bf16_t* __restrict__ h1,
                                  float* __restrict__ c, unsigned* __restrict__ sync) {
    int i = blockIdx.x * blockDim.x + threadIdx.x;
    if (i == 0) *sync = 0u;
    if (i < B_ * H_) {
        h0[i] = f2bf(0.0f);
        h1[i] = f2bf(0.0f);
        c[i]  = 0.0f;
    }
}

// ---- persistent LSTM scan -------------------------------------------------
// grid (8,32): block (mi,ni) owns output tile rows mi*16.., cols ni*16.. for
// all 1024 timesteps. 128 threads = 4 waves; wave w computes gate w via WMMA;
// the 24 weight B-fragments are re-read from global wpack at every step.
// Per step: async-stage x_t and h tiles to LDS, 24 WMMAs/wave, gate combine
// in LDS, then device-wide sync via monotonic L2 atomic counter.
__global__ __launch_bounds__(128) void lstm_persist_kernel(
    const bf16_t* __restrict__ xbf, const bf16_t* __restrict__ wpack,
    bf16_t* __restrict__ h0, bf16_t* __restrict__ h1, float* __restrict__ c,
    const float* __restrict__ bg, const float* __restrict__ bi,
    const float* __restrict__ bfv, const float* __restrict__ bo,
    unsigned* __restrict__ sync) {
    __shared__ float  tile[4][16][16];    // 4 KB gate results
    __shared__ bf16_t hsh[16][H_];        // 16 KB h tile (rows mi*16..)
    __shared__ bf16_t xsh[16][D_];        // 8 KB x_t tile

    const int wid   = threadIdx.x >> 5;   // gate id 0..3
    const int lane  = threadIdx.x & 31;
    const int lhalf = lane >> 4;          // 0 | 1
    const int l15   = lane & 15;
    const int mi = blockIdx.x;            // 0..7
    const int ni = blockIdx.y;            // 0..31

    const int ncol = ni * 16 + l15;       // hidden column for B fragments

    // B lanes: lane<16 holds K {0..7,16..23}, lane>=16 holds K {8..15,24..31}
    const bf16_t* bW  = wpack + ((size_t)wid * H_ + ncol) * KTOT + lhalf * 8;
    const bf16_t* bWh = bW + D_;
    const float* bias = (wid == 0) ? bg : (wid == 1) ? bi : (wid == 2) ? bfv : bo;
    const float  bb   = bias[ncol];

    // A lanes: lane<16 holds K {0..7,16..23}, lane>=16 holds K {8..15,24..31}
    const bf16_t* aX = &xsh[l15][0] + lhalf * 8;
    const bf16_t* aH = &hsh[l15][0] + lhalf * 8;

    for (int t = 0; t < S_; ++t) {
        const bf16_t* hin  = (t & 1) ? h1 : h0;
        bf16_t*       hout = (t & 1) ? h0 : h1;

        // ---- async-stage tiles to LDS (shared by all 4 gate-waves) ----
        {
            // h tile: 16 contiguous rows = 16384 B -> 8 iters of 128x16B
            const bf16_t* hsrc = hin + (size_t)mi * 16 * H_;
            #pragma unroll
            for (int it = 0; it < 8; ++it) {
                int e = (it * 128 + threadIdx.x) * 8;      // bf16 elements
                async_load_b128(hsrc + e, &hsh[0][0] + e);
            }
            // x tile: 16 rows, global row stride S*D -> 4 iters of 128x16B
            #pragma unroll
            for (int it = 0; it < 4; ++it) {
                int cid = it * 128 + threadIdx.x;          // 16B chunk id
                int row = cid >> 5, cc = cid & 31;
                const bf16_t* xsrc = xbf + (size_t)(mi * 16 + row) * (S_ * D_)
                                         + (size_t)t * D_ + cc * 8;
                async_load_b128(xsrc, &xsh[0][0] + row * D_ + cc * 8);
            }
            wait_async0();
            __syncthreads();
        }

        v8f acc = {};

        // x_t @ Wx^T contribution (K = 0..255), A from LDS, B from global wpack
        #pragma unroll
        for (int k = 0; k < D_; k += 32) {
            Frag a, b;
            load_frags(a, b, aX, bW, k);
            acc = __builtin_amdgcn_wmma_f32_16x16x32_bf16(false, a.v, false, b.v,
                                                          (short)0, acc, false, false);
        }
        // h @ Wh^T contribution (K = 0..511)
        #pragma unroll
        for (int k = 0; k < H_; k += 32) {
            Frag a, b;
            load_frags(a, b, aH, bWh, k);
            acc = __builtin_amdgcn_wmma_f32_16x16x32_bf16(false, a.v, false, b.v,
                                                          (short)0, acc, false, false);
        }

        // C/D layout: VGPR r -> row r + 8*lhalf, col = lane&15
        #pragma unroll
        for (int r = 0; r < 8; ++r) {
            float v = acc[r] + bb;
            v = (wid == 0) ? tanhf(v) : (1.0f / (1.0f + __expf(-v)));
            tile[wid][r + lhalf * 8][l15] = v;
        }
        __syncthreads();

        // elementwise c/h update: 256 elements, 128 threads -> 2 each
        #pragma unroll
        for (int e = 0; e < 2; ++e) {
            int idx = threadIdx.x * 2 + e;
            int m = idx >> 4, n = idx & 15;
            float g  = tile[0][m][n];
            float ii = tile[1][m][n];
            float ff = tile[2][m][n];
            float oo = tile[3][m][n];
            size_t off = (size_t)(mi * 16 + m) * H_ + (ni * 16 + n);
            float cn = g * ii + c[off] * ff;
            c[off]    = cn;
            hout[off] = f2bf(tanhf(cn) * oo);
        }

        // ---- device-wide step barrier (monotonic L2 atomic counter) ----
        __threadfence();
        __syncthreads();
        if (threadIdx.x == 0) {
            __hip_atomic_fetch_add(sync, 1u, __ATOMIC_RELEASE, __HIP_MEMORY_SCOPE_AGENT);
            const unsigned target = (unsigned)(t + 1) * NBLK;
            while (__hip_atomic_load(sync, __ATOMIC_ACQUIRE, __HIP_MEMORY_SCOPE_AGENT) < target) {
                __builtin_amdgcn_s_sleep(1);
            }
        }
        __syncthreads();
    }
}

// ---- final projection: out = h @ Wph + bp  (128x512 @ 512x10) -------------
__global__ void proj_kernel(const bf16_t* __restrict__ h, const float* __restrict__ Wph,
                            const float* __restrict__ bp, float* __restrict__ out) {
    int b = threadIdx.x;   // 0..127
    int j = blockIdx.x;    // 0..9
    float acc = 0.0f;
    for (int n = 0; n < H_; ++n)
        acc += bf2f(h[(size_t)b * H_ + n]) * Wph[(size_t)n * C_ + j];
    out[b * C_ + j] = acc + bp[j];
}

extern "C" void kernel_launch(void* const* d_in, const int* in_sizes, int n_in,
                              void* d_out, int out_size, void* d_ws, size_t ws_size,
                              hipStream_t stream) {
    const float* x   = (const float*)d_in[0];
    const float* Wgx = (const float*)d_in[1];
    const float* Wix = (const float*)d_in[2];
    const float* Wfx = (const float*)d_in[3];
    const float* Wox = (const float*)d_in[4];
    const float* Wgh = (const float*)d_in[5];
    const float* Wih = (const float*)d_in[6];
    const float* Wfh = (const float*)d_in[7];
    const float* Woh = (const float*)d_in[8];
    const float* bg  = (const float*)d_in[9];
    const float* bi  = (const float*)d_in[10];
    const float* bf  = (const float*)d_in[11];
    const float* bo  = (const float*)d_in[12];
    const float* Wph = (const float*)d_in[13];
    const float* bp  = (const float*)d_in[14];

    char* ws = (char*)d_ws;
    size_t off = 0;
    bf16_t*  xbf   = (bf16_t*)(ws + off); off += (size_t)B_ * S_ * D_ * 2;   // 64 MB
    bf16_t*  wpack = (bf16_t*)(ws + off); off += (size_t)4 * H_ * KTOT * 2;  // 3 MB
    bf16_t*  h0    = (bf16_t*)(ws + off); off += (size_t)B_ * H_ * 2;
    bf16_t*  h1    = (bf16_t*)(ws + off); off += (size_t)B_ * H_ * 2;
    float*   c     = (float*)(ws + off);  off += (size_t)B_ * H_ * 4;
    unsigned* sync = (unsigned*)(ws + off); off += 256;
    (void)in_sizes; (void)n_in; (void)out_size; (void)ws_size;

    const int nx = B_ * S_ * D_;
    conv_x_kernel<<<nx / 4 / 256, 256, 0, stream>>>(x, xbf, nx);

    const int nw = 4 * H_ * KTOT;
    pack_w_kernel<<<(nw + 255) / 256, 256, 0, stream>>>(Wgx, Wix, Wfx, Wox,
                                                        Wgh, Wih, Wfh, Woh, wpack);

    init_state_kernel<<<(B_ * H_ + 255) / 256, 256, 0, stream>>>(h0, h1, c, sync);

    lstm_persist_kernel<<<dim3(8, 32), 128, 0, stream>>>(xbf, wpack, h0, h1, c,
                                                         bg, bi, bf, bo, sync);

    // final h: written at t=1023 (odd) -> h0
    proj_kernel<<<C_, B_, 0, stream>>>(h0, Wph, bp, (float*)d_out);
}
